// CoDinoTransformer_52046413693214
// MI455X (gfx1250) — matrix operand order repaired and checked by reading, not compiled, not run on hardware
//
#include <hip/hip_runtime.h>
#include <hip/hip_bf16.h>
#include <stdint.h>

typedef __attribute__((ext_vector_type(16))) _Float16 v16h;
typedef __attribute__((ext_vector_type(8)))  float    v8f;

#define NLAYERS 6
#define BATCH   2
#define DMODEL  256
#define NHEADS  8
#define HDIM    32
#define NLVL    4
#define NPT     4
#define DFF     1024
#define S_TOTAL 20197

// ---------------------------------------------------------------------------
// CDNA5 async global->LDS copy (16B per lane), tracked by ASYNCcnt.
// vdst = LDS byte offset VGPR, vaddr = 64-bit global address VGPR pair.
// ---------------------------------------------------------------------------
__device__ __forceinline__ void async_copy_b128(uint32_t lds_off, const void* gptr) {
  unsigned long long ga = (unsigned long long)gptr;
  asm volatile("global_load_async_to_lds_b128 %0, %1, off"
               :
               : "v"(lds_off), "v"(ga)
               : "memory");
}
__device__ __forceinline__ uint32_t lds_addr_of(const void* p) {
  return (uint32_t)(size_t)p;   // low 32 bits of flat LDS address = LDS offset
}

// ---------------------------------------------------------------------------
// f32 -> f16 conversion
// ---------------------------------------------------------------------------
__global__ void cvt_f16_kernel(const float* __restrict__ in,
                               _Float16* __restrict__ out, size_t n) {
  size_t i = (size_t)blockIdx.x * blockDim.x + threadIdx.x;
  if (i < n) out[i] = (_Float16)in[i];
}

// f32 W[K,N] -> f16 Wt[N,K]  (weight transpose + convert; small matrices)
__global__ void cvt_f16_T_kernel(const float* __restrict__ in,
                                 _Float16* __restrict__ out, int K, int N) {
  int idx = blockIdx.x * 256 + threadIdx.x;
  if (idx < K * N) {
    int n = idx / K;
    int k = idx - n * K;
    out[idx] = (_Float16)in[(size_t)k * N + n];
  }
}

// ---------------------------------------------------------------------------
// elementwise add: o = a + b
// ---------------------------------------------------------------------------
__global__ void add_kernel(const float* __restrict__ a,
                           const float* __restrict__ b,
                           float* __restrict__ o, size_t n) {
  size_t i = (size_t)blockIdx.x * blockDim.x + threadIdx.x;
  if (i < n) o[i] = a[i] + b[i];
}

// ---------------------------------------------------------------------------
// WMMA GEMM: C[M,N] = A[M,K](f16) * Wt[N,K](f16, pre-transposed) + bias[N].
// Block tile 128x128, K-step 32, 256 threads = 8 waves (wave32).
// Double-buffered LDS tiles filled with global_load_async_to_lds_b128.
// Wave (wm,wn) in 4x2 grid computes a 32(M) x 64(N) sub-tile = 2x4 WMMA accs.
// ---------------------------------------------------------------------------
#define TILE_M 128
#define TILE_N 128
#define TILE_K 32

__global__ __launch_bounds__(256) void gemm_f16_wmma(
    const _Float16* __restrict__ A,   // [M,K]
    const _Float16* __restrict__ Wt,  // [N,K] (transposed weights)
    const float*    __restrict__ bias,// [N]
    float*          __restrict__ C,   // [M,N]
    int M, int N, int K, int relu)
{
  __shared__ _Float16 As[2][TILE_M][TILE_K];  // 2 x 8 KB
  __shared__ _Float16 Bs[2][TILE_N][TILE_K];  // 2 x 8 KB, [n][k]

  const int tid  = threadIdx.x;
  const int lane = tid & 31;
  const int wave = tid >> 5;     // 0..7
  const int wm   = wave >> 1;    // 0..3 -> M offset wm*32
  const int wn   = wave & 1;     // 0..1 -> N offset wn*64
  const int block_m = blockIdx.x * TILE_M;
  const int block_n = blockIdx.y * TILE_N;

  const int mrow  = lane & 15;   // A fragment row / B fragment column
  const int khalf = lane >> 4;   // lane-half selects K sub-range

  // stage one 128x32 A tile + 128x32 B tile into LDS buffer `buf`.
  // 512 chunks of 16B each per tile; 2 A chunks + 2 B chunks per thread
  // -> 4 async instructions per wave per tile.
  auto stage = [&](int buf, int k0) {
#pragma unroll
    for (int t = 0; t < 2; ++t) {
      int i    = tid + t * 256;
      int row  = i >> 2;            // 4 chunks per 32-half row
      int col8 = (i & 3) * 8;
      int gm   = block_m + row; if (gm >= M) gm = M - 1;   // clamp (store guarded)
      async_copy_b128(lds_addr_of(&As[buf][row][col8]),
                      A + (size_t)gm * K + k0 + col8);
    }
#pragma unroll
    for (int t = 0; t < 2; ++t) {
      int i    = tid + t * 256;
      int row  = i >> 2;
      int col8 = (i & 3) * 8;
      async_copy_b128(lds_addr_of(&Bs[buf][row][col8]),
                      Wt + (size_t)(block_n + row) * K + k0 + col8);
    }
  };

  v8f acc[2][4] = {};

  stage(0, 0);
  int cur = 0;
  for (int k0 = 0; k0 < K; k0 += TILE_K) {
    const int next = k0 + TILE_K;
    if (next < K) {
      stage(cur ^ 1, next);
      // previous tile's 4 async copies done; next tile's 4 still in flight
      asm volatile("s_wait_asynccnt 0x4" ::: "memory");
    } else {
      asm volatile("s_wait_asynccnt 0x0" ::: "memory");
    }
    __syncthreads();

#pragma unroll
    for (int mi = 0; mi < 2; ++mi) {
      // A fragment: lanes 0-15 hold M=0..15; VGPR v -> K = (v/4)*16 + khalf*8 + (v%4)*2
      v16h afrag;
      const int arow = wm * 32 + mi * 16 + mrow;
#pragma unroll
      for (int v = 0; v < 8; ++v) {
        int kk = (v >> 2) * 16 + khalf * 8 + (v & 3) * 2;
        afrag[2 * v]     = As[cur][arow][kk];
        afrag[2 * v + 1] = As[cur][arow][kk + 1];
      }
#pragma unroll
      for (int ni = 0; ni < 4; ++ni) {
        // B fragment: lane%16 = N col; element j -> K = (j/8)*16 + khalf*8 + j%8
        v16h bfrag;
        const int ncol  = wn * 64 + ni * 16 + mrow;
        const int kbase = khalf * 8;
#pragma unroll
        for (int j = 0; j < 16; ++j) bfrag[j] = Bs[cur][ncol][(j >> 3) * 16 + kbase + (j & 7)];
        acc[mi][ni] = __builtin_amdgcn_wmma_f32_16x16x32_f16(
            false, afrag, false, bfrag, (short)0, acc[mi][ni], false, false);
      }
    }
    __syncthreads();   // all waves done reading before buffer is overwritten
    cur ^= 1;
  }

  // ---- store: C/D layout: VGPR r -> M = r + 8*(lane>=16); N = lane%16 -----
  const int lm = lane >> 4;
  const int ln = lane & 15;
#pragma unroll
  for (int mi = 0; mi < 2; ++mi) {
#pragma unroll
    for (int ni = 0; ni < 4; ++ni) {
#pragma unroll
      for (int r = 0; r < 8; ++r) {
        int gm = block_m + wm * 32 + mi * 16 + lm * 8 + r;
        int gn = block_n + wn * 64 + ni * 16 + ln;
        if (gm < M && gn < N) {
          float v = acc[mi][ni][r] + bias[gn];
          if (relu) v = fmaxf(v, 0.0f);
          C[(size_t)gm * N + gn] = v;
        }
      }
    }
  }
}

// ---------------------------------------------------------------------------
// Multi-scale deformable attention sampling.
// One block per (batch, query): 8 waves = 8 heads, lane = channel (HD==32).
// ---------------------------------------------------------------------------
__global__ __launch_bounds__(256) void msda_sample_kernel(
    const float* __restrict__ value,   // [B*S, 256] = [B,S,NH,HD]
    const float* __restrict__ off,     // [B*S, 256] = [.., NH,NL,NP,2]
    const float* __restrict__ attnlog, // [B*S, 128] = [.., NH, 16]
    const float* __restrict__ vr,      // [B, NL, 2]
    float* __restrict__ out)           // [B*S, 256]
{
  constexpr int LVL_H[4]     = {100, 50, 25, 13};
  constexpr int LVL_W[4]     = {152, 76, 38, 19};
  constexpr int LVL_START[4] = {0, 15200, 19000, 19950};

  const int bq = blockIdx.x;          // 0 .. B*S-1
  const int b  = bq / S_TOTAL;
  const int qi = bq - b * S_TOTAL;
  const int h  = threadIdx.x >> 5;
  const int d  = threadIdx.x & 31;

  // level + base reference point of this query
  int lq = 3;
  if (qi < LVL_START[1]) lq = 0;
  else if (qi < LVL_START[2]) lq = 1;
  else if (qi < LVL_START[3]) lq = 2;
  const int rem = qi - LVL_START[lq];
  const int Wq = LVL_W[lq], Hq = LVL_H[lq];
  const int row = rem / Wq, col = rem - row * Wq;
  const float rx = (col + 0.5f) / (vr[(b * NLVL + lq) * 2 + 0] * (float)Wq);
  const float ry = (row + 0.5f) / (vr[(b * NLVL + lq) * 2 + 1] * (float)Hq);

  // softmax over 16 attention logits for this head
  const float* alog = attnlog + (size_t)bq * 128 + h * 16;
  float logits[16];
  float mx = -1e30f;
#pragma unroll
  for (int i = 0; i < 16; ++i) { logits[i] = alog[i]; mx = fmaxf(mx, logits[i]); }
  float ssum = 0.0f;
#pragma unroll
  for (int i = 0; i < 16; ++i) { logits[i] = __expf(logits[i] - mx); ssum += logits[i]; }
  const float inv = 1.0f / ssum;

  const float* offp = off + (size_t)bq * 256 + h * (NLVL * NPT * 2);

  float acc = 0.0f;
#pragma unroll
  for (int l = 0; l < NLVL; ++l) {
    const int Hl = LVL_H[l], Wl = LVL_W[l], st = LVL_START[l];
    const float refx = rx * vr[(b * NLVL + l) * 2 + 0];
    const float refy = ry * vr[(b * NLVL + l) * 2 + 1];
#pragma unroll
    for (int p = 0; p < NPT; ++p) {
      const float ox = offp[(l * NPT + p) * 2 + 0];
      const float oy = offp[(l * NPT + p) * 2 + 1];
      const float px = (refx + ox / (float)Wl) * (float)Wl - 0.5f;
      const float py = (refy + oy / (float)Hl) * (float)Hl - 0.5f;
      const float x0f = floorf(px), y0f = floorf(py);
      const float fx = px - x0f, fy = py - y0f;
      const int x0 = (int)x0f, y0 = (int)y0f;
      const float w = logits[l * NPT + p] * inv;
      float sv = 0.0f;
#pragma unroll
      for (int dy = 0; dy < 2; ++dy) {
#pragma unroll
        for (int dx = 0; dx < 2; ++dx) {
          const int xi = x0 + dx, yi = y0 + dy;
          const bool ok = (xi >= 0) && (xi < Wl) && (yi >= 0) && (yi < Hl);
          int xc = xi < 0 ? 0 : (xi > Wl - 1 ? Wl - 1 : xi);
          int yc = yi < 0 ? 0 : (yi > Hl - 1 ? Hl - 1 : yi);
          const float g =
              value[((size_t)b * S_TOTAL + st + yc * Wl + xc) * 256 + h * HDIM + d];
          const float bw = (dx ? fx : 1.0f - fx) * (dy ? fy : 1.0f - fy);
          sv += ok ? g * bw : 0.0f;
        }
      }
      acc += w * sv;
    }
  }
  out[(size_t)bq * 256 + h * HDIM + d] = acc;
}

// ---------------------------------------------------------------------------
// y = LayerNorm(x + a) * g + b ; one 256-thread block per row (D == 256)
// ---------------------------------------------------------------------------
__global__ __launch_bounds__(256) void add_ln_kernel(
    const float* __restrict__ x, const float* __restrict__ a,
    const float* __restrict__ g, const float* __restrict__ bb,
    float* __restrict__ y)
{
  __shared__ float red[256];
  const size_t base = (size_t)blockIdx.x * 256;
  const int t = threadIdx.x;

  const float v = x[base + t] + a[base + t];
  red[t] = v;
  __syncthreads();
  for (int s = 128; s > 0; s >>= 1) {
    if (t < s) red[t] += red[t + s];
    __syncthreads();
  }
  const float mean = red[0] * (1.0f / 256.0f);
  __syncthreads();

  const float dv = v - mean;
  red[t] = dv * dv;
  __syncthreads();
  for (int s = 128; s > 0; s >>= 1) {
    if (t < s) red[t] += red[t + s];
    __syncthreads();
  }
  const float var = red[0] * (1.0f / 256.0f);
  const float r = rsqrtf(var + 1e-5f);
  y[base + t] = dv * r * g[t] + bb[t];
}

// ---------------------------------------------------------------------------
// host driver
// ---------------------------------------------------------------------------
extern "C" void kernel_launch(void* const* d_in, const int* in_sizes, int n_in,
                              void* d_out, int out_size, void* d_ws, size_t ws_size,
                              hipStream_t stream) {
  (void)in_sizes; (void)n_in; (void)out_size; (void)ws_size;

  const float* src  = (const float*)d_in[0];
  const float* pos  = (const float*)d_in[1];
  const float* vr   = (const float*)d_in[2];
  const float* Wv   = (const float*)d_in[3];
  const float* bv   = (const float*)d_in[4];
  const float* Woff = (const float*)d_in[5];
  const float* boff = (const float*)d_in[6];
  const float* Wa   = (const float*)d_in[7];
  const float* ba   = (const float*)d_in[8];
  const float* Wo   = (const float*)d_in[9];
  const float* bo   = (const float*)d_in[10];
  const float* g1   = (const float*)d_in[11];
  const float* be1  = (const float*)d_in[12];
  const float* Wf1  = (const float*)d_in[13];
  const float* bf1  = (const float*)d_in[14];
  const float* Wf2  = (const float*)d_in[15];
  const float* bf2  = (const float*)d_in[16];
  const float* g2   = (const float*)d_in[17];
  const float* be2  = (const float*)d_in[18];

  const int M = BATCH * S_TOTAL;   // 40394 rows

  // ---- workspace carve-up ------------------------------------------------
  char*  ws   = (char*)d_ws;
  size_t woff = 0;
  auto alloc = [&](size_t bytes) -> void* {
    void* p = ws + woff;
    woff += (bytes + 255) & ~(size_t)255;
    return p;
  };
  float*    x     = (float*)alloc((size_t)M * 256 * 4);
  float*    qf    = (float*)alloc((size_t)M * 256 * 4);
  float*    val   = (float*)alloc((size_t)M * 256 * 4);
  float*    offb  = (float*)alloc((size_t)M * 256 * 4);
  float*    attn  = (float*)alloc((size_t)M * 128 * 4);
  float*    samp  = (float*)alloc((size_t)M * 256 * 4);
  float*    tmp   = (float*)alloc((size_t)M * 256 * 4);
  float*    hid   = (float*)alloc((size_t)M * DFF * 4);
  _Float16* actbf = (_Float16*)alloc((size_t)M * DFF * 2);   // reused f16 acts
  _Float16* wbf   = (_Float16*)alloc((size_t)DMODEL * DFF * 2); // transposed f16 W

  hipMemcpyAsync(x, src, (size_t)M * 256 * 4, hipMemcpyDeviceToDevice, stream);

  auto cvt = [&](const float* in, _Float16* out, size_t n) {
    cvt_f16_kernel<<<(unsigned)((n + 255) / 256), 256, 0, stream>>>(in, out, n);
  };
  auto cvtT = [&](const float* in, _Float16* out, int K, int N) {
    cvt_f16_T_kernel<<<(unsigned)(((size_t)K * N + 255) / 256), 256, 0, stream>>>(
        in, out, K, N);
  };
  auto gemm = [&](const _Float16* A, const _Float16* Bt, const float* bias,
                  float* Cm, int Mr, int Nc, int Kc, int relu) {
    dim3 grid((Mr + TILE_M - 1) / TILE_M, (Nc + TILE_N - 1) / TILE_N);
    gemm_f16_wmma<<<grid, 256, 0, stream>>>(A, Bt, bias, Cm, Mr, Nc, Kc, relu);
  };

  for (int i = 0; i < NLAYERS; ++i) {
    // q = x + pos
    add_kernel<<<(unsigned)(((size_t)M * 256 + 255) / 256), 256, 0, stream>>>(
        x, pos, qf, (size_t)M * 256);

    // value = x @ Wv + bv
    cvt(x, actbf, (size_t)M * 256);
    cvtT(Wv + (size_t)i * 256 * 256, wbf, 256, 256);
    gemm(actbf, wbf, bv + i * 256, val, M, 256, 256, 0);

    // offsets + attention logits from q
    cvt(qf, actbf, (size_t)M * 256);
    cvtT(Woff + (size_t)i * 256 * 256, wbf, 256, 256);
    gemm(actbf, wbf, boff + i * 256, offb, M, 256, 256, 0);
    cvtT(Wa + (size_t)i * 256 * 128, wbf, 256, 128);
    gemm(actbf, wbf, ba + i * 128, attn, M, 128, 256, 0);

    // deformable sampling
    msda_sample_kernel<<<M, 256, 0, stream>>>(val, offb, attn, vr, samp);

    // out projection
    cvt(samp, actbf, (size_t)M * 256);
    cvtT(Wo + (size_t)i * 256 * 256, wbf, 256, 256);
    gemm(actbf, wbf, bo + i * 256, tmp, M, 256, 256, 0);

    // x = LN(x + attn_out)
    add_ln_kernel<<<M, 256, 0, stream>>>(x, tmp, g1 + i * 256, be1 + i * 256, x);

    // FFN
    cvt(x, actbf, (size_t)M * 256);
    cvtT(Wf1 + (size_t)i * 256 * DFF, wbf, 256, DFF);
    gemm(actbf, wbf, bf1 + i * DFF, hid, M, DFF, 256, 1);
    cvt(hid, actbf, (size_t)M * DFF);
    cvtT(Wf2 + (size_t)i * DFF * 256, wbf, DFF, 256);
    gemm(actbf, wbf, bf2 + i * 256, tmp, M, 256, DFF, 0);

    // x = LN(x + ffn_out) ; last layer writes straight to d_out
    float* dst = (i == NLAYERS - 1) ? (float*)d_out : x;
    add_ln_kernel<<<M, 256, 0, stream>>>(x, tmp, g2 + i * 256, be2 + i * 256, dst);
  }
}
